// BaseAttention_81329500717641
// MI455X (gfx1250) — matrix operand order repaired and checked by reading, not compiled, not run on hardware
//
#include <hip/hip_runtime.h>
#include <hip/hip_bf16.h>

typedef __attribute__((ext_vector_type(16))) _Float16 v16h;
typedef __attribute__((ext_vector_type(8)))  _Float16 v8h;
typedef __attribute__((ext_vector_type(8)))  float    v8f;

__device__ __forceinline__ v16h combine16(v8h lo, v8h hi) {
  v16h r;
#pragma unroll
  for (int i = 0; i < 8; ++i) { r[i] = lo[i]; r[8 + i] = hi[i]; }
  return r;
}

__device__ __forceinline__ v8f zero8() {
  v8f z;
#pragma unroll
  for (int i = 0; i < 8; ++i) z[i] = 0.0f;
  return z;
}

__device__ __forceinline__ v8f wmma_f16(v16h a, v16h b, v8f c) {
  // D = A(16x32 f16) x B(32x16 f16) + C(16x16 f32)
  return __builtin_amdgcn_wmma_f32_16x16x32_f16(
      /*neg_a=*/false, a, /*neg_b=*/false, b,
      /*c_mod=*/(short)0, c, /*reuse_a=*/false, /*reuse_b=*/false);
}

// Per-lane async copy of 16B global -> LDS (VGLOBAL GLOBAL_LOAD_ASYNC_TO_LDS_B128,
// tracked by ASYNCcnt). ldsoff is the wave-relative LDS byte offset (low 32 bits
// of the generic address of a __shared__ object).
__device__ __forceinline__ void async_copy_b128(unsigned ldsoff, const void* gaddr) {
  asm volatile("global_load_async_to_lds_b128 %0, %1, off"
               :: "v"(ldsoff), "v"((unsigned long long)gaddr)
               : "memory");
}
__device__ __forceinline__ void wait_asynccnt0() {
  asm volatile("s_wait_asynccnt 0x0" ::: "memory");
}
__device__ __forceinline__ unsigned lds_offset_of(const void* p) {
  return (unsigned)(unsigned long long)p; // low 32 bits of generic LDS addr = LDS offset
}

// ---------------------------------------------------------------------------
// Generic tiled GEMM:  D[M,N] = A[M,K] * B[K,N] (+ bias[N])
// Block: 256 threads = 8 waves. Tile: BM=64, BN=128, BK=32.
// Waves 2x4; each wave owns a 32x32 sub-tile = 2x2 WMMA accumulators.
// A staged to LDS as f16 [64][40]; B staged TRANSPOSED as f16 [128][40]
// so that B-fragments (K contiguous per column) are two 16B LDS reads per lane.
// ---------------------------------------------------------------------------
template <bool A_IS_HALF, bool BIAS, bool OUT_HALF>
__global__ __launch_bounds__(256) void gemm_wmma(
    const void* __restrict__ Aptr, const float* __restrict__ B,
    const float* __restrict__ bias, void* __restrict__ Dptr,
    int M, int N, int K) {
  __shared__ __align__(16) _Float16 lds_a[64][40];   // [m][k], pad 8
  __shared__ __align__(16) _Float16 lds_bt[128][40]; // [n][k], pad 8

  const int m0 = blockIdx.y * 64;
  const int n0 = blockIdx.x * 128;
  const int wave = threadIdx.x >> 5;
  const int lane = threadIdx.x & 31;
  const int l = lane & 15;
  const int hi = lane >> 4;
  const int wm = wave >> 2; // 0..1
  const int wn = wave & 3;  // 0..3

  v8f acc[2][2];
#pragma unroll
  for (int i = 0; i < 2; ++i)
#pragma unroll
    for (int j = 0; j < 2; ++j) acc[i][j] = zero8();

  for (int k0 = 0; k0 < K; k0 += 32) {
    // prefetch next K-tile (one cacheline-touch per thread for A and B streams)
    if (k0 + 32 < K) {
      int ra = threadIdx.x >> 5, ca = threadIdx.x & 31;
      if (A_IS_HALF)
        __builtin_prefetch(((const _Float16*)Aptr) + (size_t)(m0 + ra) * K + k0 + 32 + ca, 0, 1);
      else
        __builtin_prefetch(((const float*)Aptr) + (size_t)(m0 + ra) * K + k0 + 32 + ca, 0, 1);
      int kb = threadIdx.x >> 7, nb = threadIdx.x & 127;
      __builtin_prefetch(B + (size_t)(k0 + 32 + kb) * N + n0 + nb, 0, 1);
    }
    __syncthreads();
    // stage A tile 64x32 -> f16
#pragma unroll
    for (int i = threadIdx.x; i < 64 * 32; i += 256) {
      int r = i >> 5, c = i & 31;
      if (A_IS_HALF) {
        lds_a[r][c] = ((const _Float16*)Aptr)[(size_t)(m0 + r) * K + k0 + c];
      } else {
        lds_a[r][c] = (_Float16)((const float*)Aptr)[(size_t)(m0 + r) * K + k0 + c];
      }
    }
    // stage B tile 32x128 -> f16, transposed into [n][k]
#pragma unroll
    for (int i = threadIdx.x; i < 32 * 128; i += 256) {
      int kk = i >> 7, n = i & 127;
      lds_bt[n][kk] = (_Float16)B[(size_t)(k0 + kk) * N + n0 + n];
    }
    __syncthreads();

    v16h afrag[2], bfrag[2];
#pragma unroll
    for (int mt = 0; mt < 2; ++mt) {
      const _Float16* ar = &lds_a[wm * 32 + mt * 16 + l][0];
      afrag[mt] = combine16(*(const v8h*)(ar + hi * 8),
                            *(const v8h*)(ar + 16 + hi * 8));
    }
#pragma unroll
    for (int nt = 0; nt < 2; ++nt) {
      const _Float16* br = &lds_bt[wn * 32 + nt * 16 + l][hi * 8];
      bfrag[nt] = combine16(*(const v8h*)br, *(const v8h*)(br + 16));
    }
#pragma unroll
    for (int mt = 0; mt < 2; ++mt)
#pragma unroll
      for (int nt = 0; nt < 2; ++nt)
        acc[mt][nt] = wmma_f16(afrag[mt], bfrag[nt], acc[mt][nt]);
  }

  // epilogue: D layout lane(l,hi), VGPR r -> M = r + 8*hi, N = l
#pragma unroll
  for (int mt = 0; mt < 2; ++mt)
#pragma unroll
    for (int nt = 0; nt < 2; ++nt)
#pragma unroll
      for (int r = 0; r < 8; ++r) {
        int row = m0 + wm * 32 + mt * 16 + r + 8 * hi;
        int col = n0 + wn * 32 + nt * 16 + l;
        float v = acc[mt][nt][r];
        if (BIAS) v += bias[col];
        if (OUT_HALF)
          ((_Float16*)Dptr)[(size_t)row * N + col] = (_Float16)v;
        else
          ((float*)Dptr)[(size_t)row * N + col] = v;
      }
}

// ---------------------------------------------------------------------------
// RoPE + head rearrange: qkv f16 [B*S][6144] -> Qr/Kr/Vr f16 [b*16+h][s][128]
// out[d] = x[d]*cos + (d<64 ? -x[d+64] : x[d-64]) * sin ; freq = 1e4^-( (d%64)/64 )
// ---------------------------------------------------------------------------
__global__ __launch_bounds__(256) void rope_kernel(
    const _Float16* __restrict__ qkv, const int* __restrict__ pos_ids,
    _Float16* __restrict__ Qr, _Float16* __restrict__ Kr,
    _Float16* __restrict__ Vr) {
  const int token = blockIdx.x;           // b*2048 + s
  const int b = token >> 11;
  const int s = token & 2047;
  const float pos = (float)pos_ids[token];
  const _Float16* base = qkv + (size_t)token * 6144;
  for (int i = threadIdx.x; i < 2048; i += 256) {
    int h = i >> 7, d = i & 127;
    int dh = d & 63;
    // inv_freq = 10000^-(dh/64) = exp(-ln(1e4)/64 * dh)
    float inv = __expf(-0.14391156831212787f * (float)dh);
    float ang = pos * inv;
    float sn, cs;
    __sincosf(ang, &sn, &cs);
    int pd = (d < 64) ? (i + 64) : (i - 64);
    float sign = (d < 64) ? -1.0f : 1.0f;
    float q = (float)base[i],         qp = (float)base[pd];
    float k = (float)base[2048 + i],  kp = (float)base[2048 + pd];
    float v = (float)base[4096 + i];
    size_t off = (((size_t)(b * 16 + h) * 2048) + s) * 128 + d;
    Qr[off] = (_Float16)(q * cs + sign * qp * sn);
    Kr[off] = (_Float16)(k * cs + sign * kp * sn);
    Vr[off] = (_Float16)v;
  }
}

// ---------------------------------------------------------------------------
// Flash attention, causal. Block = 128 threads = 4 waves; one 16-row Q tile
// per wave; K/V staged in LDS (K via GLOBAL_LOAD_ASYNC_TO_LDS_B128 / ASYNCcnt,
// V transposed via VGPR path) shared by the 4 waves.
// Online softmax: per-row stats held per-lane (rows r+8*hi), 16-lane shfl
// reductions. P goes through wave-private LDS to become an A-fragment.
// ---------------------------------------------------------------------------
__global__ __launch_bounds__(128) void flash_attn(
    const _Float16* __restrict__ Qr, const _Float16* __restrict__ Kr,
    const _Float16* __restrict__ Vr, _Float16* __restrict__ attn) {
  __shared__ __align__(16) _Float16 lds_k[32][136];  // [kv][d]
  __shared__ __align__(16) _Float16 lds_vt[128][40]; // [d][kv]
  __shared__ __align__(16) _Float16 lds_p[4][16][32]; // per-wave P scratch

  const int bh = blockIdx.y;       // b*16 + h
  const int b = bh >> 4, h = bh & 15;
  const int wave = threadIdx.x >> 5;
  const int lane = threadIdx.x & 31;
  const int l = lane & 15;
  const int hi = lane >> 4;
  const int qt = blockIdx.x * 4 + wave; // Q tile index, 0..127
  const size_t bhoff = (size_t)bh * 2048 * 128;
  const float sc = 0.08838834764831845f; // 1/sqrt(128)

  // Load Q fragments for all 4 K-steps of head_dim=128 (A 16x32 layout).
  v16h qf[4];
#pragma unroll
  for (int ks = 0; ks < 4; ++ks) {
    const _Float16* qp = Qr + bhoff + (size_t)(qt * 16 + l) * 128 + ks * 32 + hi * 8;
    qf[ks] = combine16(*(const v8h*)qp, *(const v8h*)(qp + 16));
  }

  float m[8], lsum[8];
  v8f acc[8];
#pragma unroll
  for (int r = 0; r < 8; ++r) { m[r] = -3.0e38f; lsum[r] = 0.0f; }
#pragma unroll
  for (int dt = 0; dt < 8; ++dt) acc[dt] = zero8();

  const int nkv_block = blockIdx.x * 2 + 2; // covers max q row in this block
  const int nkv_w = qt / 2 + 1;             // this wave's causal KV-tile limit

  for (int kvt = 0; kvt < nkv_block; ++kvt) {
    // prefetch next KV tile (8KB each; 128 threads x 64B)
    if (kvt + 1 < nkv_block) {
      const char* pk = (const char*)(Kr + bhoff + (size_t)((kvt + 1) * 32) * 128)
                       + (size_t)threadIdx.x * 64;
      const char* pv = (const char*)(Vr + bhoff + (size_t)((kvt + 1) * 32) * 128)
                       + (size_t)threadIdx.x * 64;
      __builtin_prefetch(pk, 0, 1);
      __builtin_prefetch(pv, 0, 1);
    }
    __syncthreads();
    // stage K tile 32x128 (natural layout): async L2->LDS, 16B per lane per op
#pragma unroll
    for (int j = 0; j < 4; ++j) {
      int idx = threadIdx.x + j * 128;  // 0..511 chunks of 8 halves
      int r = idx >> 4, cg = idx & 15;
      async_copy_b128(lds_offset_of(&lds_k[r][cg * 8]),
                      Kr + bhoff + (size_t)(kvt * 32 + r) * 128 + cg * 8);
    }
    // stage V tile transposed: [d][kv] (VGPR path, does the transpose)
#pragma unroll
    for (int j = 0; j < 4; ++j) {
      int idx = threadIdx.x + j * 128;
      int r = idx >> 4, cg = idx & 15;
      v8h vv = *(const v8h*)(Vr + bhoff + (size_t)(kvt * 32 + r) * 128 + cg * 8);
#pragma unroll
      for (int e = 0; e < 8; ++e) lds_vt[cg * 8 + e][r] = vv[e];
    }
    wait_asynccnt0();  // K async copies landed in LDS (barrier doesn't cover ASYNCcnt)
    __syncthreads();

    if (kvt < nkv_w) {
      // S = Q * K^T for 32 KV columns -> two 16x16 f32 tiles
      v8f s0 = zero8(), s1 = zero8();
#pragma unroll
      for (int ks = 0; ks < 4; ++ks) {
        const _Float16* kp0 = &lds_k[l][ks * 32 + hi * 8];
        const _Float16* kp1 = &lds_k[16 + l][ks * 32 + hi * 8];
        v16h bf0 = combine16(*(const v8h*)kp0, *(const v8h*)(kp0 + 16));
        v16h bf1 = combine16(*(const v8h*)kp1, *(const v8h*)(kp1 + 16));
        s0 = wmma_f16(qf[ks], bf0, s0);
        s1 = wmma_f16(qf[ks], bf1, s1);
      }

      const int kv0 = kvt * 32 + l;
      const int kv1 = kv0 + 16;
      float p0[8], p1[8], alpha[8];
#pragma unroll
      for (int r = 0; r < 8; ++r) {
        int qrow = qt * 16 + r + 8 * hi;
        float v0 = (kv0 <= qrow) ? s0[r] * sc : -3.0e38f;
        float v1 = (kv1 <= qrow) ? s1[r] * sc : -3.0e38f;
        float t = fmaxf(v0, v1);
#pragma unroll
        for (int msk = 1; msk < 16; msk <<= 1) t = fmaxf(t, __shfl_xor(t, msk, 32));
        float mn = fmaxf(m[r], t);
        float a = __expf(m[r] - mn);
        float e0 = __expf(v0 - mn);
        float e1 = __expf(v1 - mn);
        float rs = e0 + e1;
#pragma unroll
        for (int msk = 1; msk < 16; msk <<= 1) rs += __shfl_xor(rs, msk, 32);
        lsum[r] = lsum[r] * a + rs;
        m[r] = mn;
        alpha[r] = a;
        p0[r] = e0;
        p1[r] = e1;
      }
#pragma unroll
      for (int dt = 0; dt < 8; ++dt)
#pragma unroll
        for (int r = 0; r < 8; ++r) acc[dt][r] = acc[dt][r] * alpha[r];

      // redistribute P (D layout) into A-fragment layout via wave-private LDS
#pragma unroll
      for (int r = 0; r < 8; ++r) {
        lds_p[wave][r + 8 * hi][l] = (_Float16)p0[r];
        lds_p[wave][r + 8 * hi][16 + l] = (_Float16)p1[r];
      }
      const _Float16* pr = &lds_p[wave][l][0];
      v16h pf = combine16(*(const v8h*)(pr + hi * 8),
                          *(const v8h*)(pr + 16 + hi * 8));
      // acc += P * V  (K=32 over the KV tile, 8 column tiles of head dim)
#pragma unroll
      for (int dt = 0; dt < 8; ++dt) {
        const _Float16* vp = &lds_vt[dt * 16 + l][hi * 8];
        v16h vf = combine16(*(const v8h*)vp, *(const v8h*)(vp + 16));
        acc[dt] = wmma_f16(pf, vf, acc[dt]);
      }
    }
  }

  // epilogue: attn[b][s][h*128 + d], f16
  float inv[8];
#pragma unroll
  for (int r = 0; r < 8; ++r) inv[r] = 1.0f / lsum[r];
#pragma unroll
  for (int dt = 0; dt < 8; ++dt)
#pragma unroll
    for (int r = 0; r < 8; ++r) {
      int qrow = qt * 16 + r + 8 * hi;
      int d = dt * 16 + l;
      attn[((size_t)b * 2048 + qrow) * 2048 + h * 128 + d] =
          (_Float16)(acc[dt][r] * inv[r]);
    }
}

// ---------------------------------------------------------------------------
extern "C" void kernel_launch(void* const* d_in, const int* in_sizes, int n_in,
                              void* d_out, int out_size, void* d_ws, size_t ws_size,
                              hipStream_t stream) {
  const float* hs    = (const float*)d_in[0];  // [4,2048,2048]
  const int*   pos   = (const int*)d_in[1];    // [4,2048]
  const float* qkv_w = (const float*)d_in[2];  // [2048,6144]
  const float* qkv_b = (const float*)d_in[3];  // [6144]
  const float* o_w   = (const float*)d_in[4];  // [2048,2048]
  float* out = (float*)d_out;                  // [4,2048,2048] f32

  char* ws = (char*)d_ws;
  _Float16* qkv = (_Float16*)ws;                                    // 96 MB
  _Float16* Qr  = (_Float16*)(ws + (size_t)96  * 1024 * 1024);      // 32 MB
  _Float16* Kr  = (_Float16*)(ws + (size_t)128 * 1024 * 1024);      // 32 MB
  _Float16* Vr  = (_Float16*)(ws + (size_t)160 * 1024 * 1024);      // 32 MB
  _Float16* attn = qkv; // reuse qkv region after RoPE consumed it

  // 1) QKV projection: [8192,2048] x [2048,6144] + bias -> f16
  gemm_wmma<false, true, true>
      <<<dim3(6144 / 128, 8192 / 64), 256, 0, stream>>>(
          hs, qkv_w, qkv_b, qkv, 8192, 6144, 2048);

  // 2) RoPE + rearrange to [b*16+h][s][128]
  rope_kernel<<<8192, 256, 0, stream>>>(qkv, pos, Qr, Kr, Vr);

  // 3) causal flash attention -> attn f16 [8192][2048]
  flash_attn<<<dim3(32, 64), 128, 0, stream>>>(Qr, Kr, Vr, attn);

  // 4) output projection: [8192,2048] x [2048,2048] -> f32 d_out
  gemm_wmma<true, false, false>
      <<<dim3(2048 / 128, 8192 / 64), 256, 0, stream>>>(
          attn, o_w, nullptr, out, 8192, 2048, 2048);
}
